// JointBetaCVAE_2585570312224
// MI455X (gfx1250) — matrix operand order repaired and checked by reading, not compiled, not run on hardware
//
#include <hip/hip_runtime.h>
#include <hip/hip_bf16.h>
#include <math.h>

typedef _Float16 half16 __attribute__((ext_vector_type(16)));
typedef float    float8 __attribute__((ext_vector_type(8)));

static constexpr int BB  = 256;   // batch
static constexpr int PP  = 64;    // persons per batch
static constexpr int HH  = 64;    // hidden
static constexpr int NDp = 8;
static constexpr int NN  = BB * PP;  // 16384

// ---------------------------------------------------------------------------
// Async global->LDS copy (CDNA5: GLOBAL_LOAD_ASYNC_TO_LDS_B128, ASYNCcnt)
// lds_off = LDS byte offset (low 32 bits of a generic pointer to __shared__)
// ---------------------------------------------------------------------------
__device__ __forceinline__ void async_g2l_b128(unsigned lds_off, const void* gaddr)
{
    asm volatile("global_load_async_to_lds_b128 %0, %1, off"
                 :: "v"(lds_off), "v"(gaddr)
                 : "memory");
}
__device__ __forceinline__ void wait_asynccnt0()
{
    asm volatile("s_wait_asynccnt 0x0" ::: "memory");
}

// ---------------------------------------------------------------------------
// K0: convert W1 (200x128, pad to 224x128) and W2 (128x16) to f16 in workspace
// ---------------------------------------------------------------------------
__global__ __launch_bounds__(256)
void wconv_kernel(const float* __restrict__ W1, const float* __restrict__ W2,
                  _Float16* __restrict__ W1h, _Float16* __restrict__ W2h)
{
    for (int idx = threadIdx.x + blockIdx.x * 256; idx < 224 * 128;
         idx += gridDim.x * 256) {
        int k = idx >> 7;
        W1h[idx] = (_Float16)((k < 200) ? W1[idx] : 0.f);
    }
    for (int idx = threadIdx.x + blockIdx.x * 256; idx < 128 * 16;
         idx += gridDim.x * 256) {
        W2h[idx] = (_Float16)W2[idx];
    }
}

// ---------------------------------------------------------------------------
// K1: three GEMMs Y = x_enc @ W + bias  (16384x64 @ 64x64), WMMA f16->f32
//     blockIdx.x = row tile (1024), blockIdx.y = which matrix (3)
//     4 waves/block, wave = column tile
// ---------------------------------------------------------------------------
__global__ __launch_bounds__(128)
void gemm3_kernel(const float* __restrict__ X,
                  const float* __restrict__ Wa, const float* __restrict__ ba, float* __restrict__ Ya,
                  const float* __restrict__ Wb, const float* __restrict__ bb, float* __restrict__ Yb,
                  const float* __restrict__ Wc, const float* __restrict__ bc, float* __restrict__ Yc)
{
    int rowTile = blockIdx.x;
    int which   = blockIdx.y;
    const float* W    = (which == 0) ? Wa : (which == 1) ? Wb : Wc;
    const float* bias = (which == 0) ? ba : (which == 1) ? bb : bc;
    float*       Y    = (which == 0) ? Ya : (which == 1) ? Yb : Yc;

    int wave = threadIdx.x >> 5;
    int lane = threadIdx.x & 31;
    int m  = lane & 15;
    int kg = lane >> 4;

    int row = rowTile * 16 + m;
    const float* xr = X + (size_t)row * 64;

    half16 a0, a1;
#pragma unroll
    for (int t = 0; t < 8; ++t) {
        a0[t]     = (_Float16)xr[kg * 16 + t];
        a0[8 + t] = (_Float16)xr[kg * 16 + 8 + t];
        a1[t]     = (_Float16)xr[32 + kg * 16 + t];
        a1[8 + t] = (_Float16)xr[32 + kg * 16 + 8 + t];
    }
    int n = wave * 16 + m;
    half16 b0, b1;
#pragma unroll
    for (int t = 0; t < 16; ++t) {
        b0[t] = (_Float16)W[(kg * 16 + t) * 64 + n];
        b1[t] = (_Float16)W[(32 + kg * 16 + t) * 64 + n];
    }
    float8 c = {};
    c = __builtin_amdgcn_wmma_f32_16x16x32_f16(false, a0, false, b0, (short)0, c, false, false);
    c = __builtin_amdgcn_wmma_f32_16x16x32_f16(false, a1, false, b1, (short)0, c, false, false);

    float bv = bias[n];
#pragma unroll
    for (int v = 0; v < 8; ++v) {
        int rout = rowTile * 16 + v + 8 * kg;
        Y[(size_t)rout * 64 + n] = c[v] + bv;
    }
}

// ---------------------------------------------------------------------------
// K2: LZ = xl@Wl_z ; T = EZ + LZ + bl_z ; key_mask = (sum|xe| > 0)
// ---------------------------------------------------------------------------
__global__ __launch_bounds__(64)
void prep_kernel(const float* __restrict__ x_enc, const float* __restrict__ x_last,
                 const float* __restrict__ Wl_z, const float* __restrict__ bl_z,
                 const float* __restrict__ EZ, float* __restrict__ LZ,
                 float* __restrict__ T, float* __restrict__ keymask)
{
    int n = blockIdx.x;
    int a = threadIdx.x;
    float x0 = x_last[n * 2 + 0], x1 = x_last[n * 2 + 1];
    float lz = x0 * Wl_z[a] + x1 * Wl_z[64 + a];
    LZ[(size_t)n * 64 + a] = lz;
    T[(size_t)n * 64 + a]  = EZ[(size_t)n * 64 + a] + lz + bl_z[a];

    __shared__ float red[64];
    red[a] = fabsf(x_enc[(size_t)n * 64 + a]);
    __syncthreads();
    if (a == 0) {
        float s = 0.f;
        for (int i = 0; i < 64; ++i) s += red[i];
        keymask[n] = (s > 0.f) ? 1.f : 0.f;
    }
}

// ---------------------------------------------------------------------------
// K3: social-attention scores + masked softmax -> alpha[b,i,j]
//     one wave per (b,i); each lane owns ATT components lane, lane+32
// ---------------------------------------------------------------------------
__global__ __launch_bounds__(32)
void score_kernel(const float* __restrict__ E, const float* __restrict__ C,
                  const float* __restrict__ x_last,
                  const float* __restrict__ Wl_x, const float* __restrict__ bl_x,
                  const float* __restrict__ Wf_x, const float* __restrict__ bf_x,
                  const float* __restrict__ keymask, float* __restrict__ alpha)
{
    int bi = blockIdx.x;          // b*64 + i
    int b = bi >> 6, i = bi & 63;
    int lane = threadIdx.x;
    int a0 = lane, a1 = lane + 32;
    int ri = b * 64 + i;

    float ci0 = C[(size_t)ri * 64 + a0], ci1 = C[(size_t)ri * 64 + a1];
    float xli0 = x_last[ri * 2], xli1 = x_last[ri * 2 + 1];
    float wl00 = Wl_x[a0], wl01 = Wl_x[a1];
    float wl10 = Wl_x[64 + a0], wl11 = Wl_x[64 + a1];
    float bl0 = bl_x[a0], bl1 = bl_x[a1];
    float wf0 = Wf_x[a0], wf1 = Wf_x[a1];
    float bf = bf_x[0];

    __shared__ float sc[64];
    for (int j = 0; j < 64; ++j) {
        int rj = b * 64 + j;
        if (j < 63)  // prefetch next key row into cache (global_prefetch_b8)
            __builtin_prefetch(&E[(size_t)(rj + 1) * 64 + a0], 0, 1);
        float dx = x_last[rj * 2] - xli0;
        float dy = x_last[rj * 2 + 1] - xli1;
        float v0 = tanhf(E[(size_t)rj * 64 + a0] + dx * wl00 + dy * wl10 + bl0 + ci0);
        float v1 = tanhf(E[(size_t)rj * 64 + a1] + dx * wl01 + dy * wl11 + bl1 + ci1);
        float part = v0 * wf0 + v1 * wf1;
#pragma unroll
        for (int o = 16; o > 0; o >>= 1) part += __shfl_xor(part, o, 32);
        if (lane == 0) sc[j] = part + bf;
    }
    __syncthreads();

    int j0 = lane, j1 = lane + 32;
    float m0 = keymask[b * 64 + j0], m1 = keymask[b * 64 + j1];
    float s0 = (m0 > 0.f) ? sc[j0] : -1e9f;
    float s1 = (m1 > 0.f) ? sc[j1] : -1e9f;
    float mx = fmaxf(s0, s1);
#pragma unroll
    for (int o = 16; o > 0; o >>= 1) mx = fmaxf(mx, __shfl_xor(mx, o, 32));
    float e0 = (m0 > 0.f) ? expf(s0 - mx) : 0.f;
    float e1 = (m1 > 0.f) ? expf(s1 - mx) : 0.f;
    float sum = e0 + e1;
#pragma unroll
    for (int o = 16; o > 0; o >>= 1) sum += __shfl_xor(sum, o, 32);
    float inv = 1.f / (sum + 1e-10f);
    alpha[(size_t)ri * 64 + j0] = e0 * inv;
    alpha[(size_t)ri * 64 + j1] = e1 * inv;
}

// ---------------------------------------------------------------------------
// K4: social = alpha(64x64) @ xe(64x64) per batch.
//     Stage both operands via async global->LDS (b128/lane), then WMMA with
//     on-the-fly f32->f16 conversion of the LDS tiles.
// ---------------------------------------------------------------------------
__global__ __launch_bounds__(256)
void social_kernel(const float* __restrict__ alpha, const float* __restrict__ x_enc,
                   float* __restrict__ social)
{
    int b = blockIdx.x;
    __shared__ __align__(16) float af[64 * 64];
    __shared__ __align__(16) float xf[64 * 64];
    {
        int t = threadIdx.x;                       // each thread copies 16 floats
        const float* ga = alpha + (size_t)b * 4096 + t * 16;
        const float* gx = x_enc + (size_t)b * 4096 + t * 16;
        unsigned la = (unsigned)(unsigned long long)&af[t * 16];
        unsigned lx = (unsigned)(unsigned long long)&xf[t * 16];
#pragma unroll
        for (int k = 0; k < 4; ++k) {
            async_g2l_b128(la + 16u * k, ga + 4 * k);
            async_g2l_b128(lx + 16u * k, gx + 4 * k);
        }
    }
    wait_asynccnt0();
    __syncthreads();

    int wave = threadIdx.x >> 5, lane = threadIdx.x & 31;
    int m = lane & 15, kg = lane >> 4;
    for (int tile = wave; tile < 16; tile += 8) {
        int tr = tile >> 2, tc = tile & 3;
        float8 c = {};
#pragma unroll
        for (int kb = 0; kb < 64; kb += 32) {
            half16 av, bv;
#pragma unroll
            for (int t = 0; t < 8; ++t) {
                av[t]     = (_Float16)af[(tr * 16 + m) * 64 + kb + kg * 16 + t];
                av[8 + t] = (_Float16)af[(tr * 16 + m) * 64 + kb + kg * 16 + 8 + t];
            }
#pragma unroll
            for (int t = 0; t < 16; ++t)
                bv[t] = (_Float16)xf[(kb + kg * 16 + t) * 64 + tc * 16 + m];
            c = __builtin_amdgcn_wmma_f32_16x16x32_f16(false, av, false, bv, (short)0, c, false, false);
        }
#pragma unroll
        for (int v = 0; v < 8; ++v)
            social[((size_t)b * 64 + tr * 16 + v + 8 * kg) * 64 + tc * 16 + m] = c[v];
    }
}

// ---------------------------------------------------------------------------
// K5: sequential scan over j=0..63. One block = 16 batch rows, 8 waves.
//     Per step: s -> masked softmax -> z_prev -> MLP via WMMA -> z
//     LZ[:, j, :] staged via async global->LDS each step.
// ---------------------------------------------------------------------------
__global__ __launch_bounds__(256)
void scan_kernel(const float* __restrict__ x_enc, const float* __restrict__ eps,
                 const float* __restrict__ T, const float* __restrict__ LZ,
                 const float* __restrict__ Wf_z, const float* __restrict__ bf_z,
                 const float* __restrict__ social,
                 const _Float16* __restrict__ W1h, const float* __restrict__ b1,
                 const _Float16* __restrict__ W2h, const float* __restrict__ b2,
                 float* __restrict__ out_means, float* __restrict__ out_logs,
                 float* __restrict__ out_zs)
{
    int bb0 = blockIdx.x * 16;
    int tid = threadIdx.x;

    __shared__ float    zsb[16][64][8];               // running latent buffer
    __shared__ _Float16 fullh[16][224];               // MLP input (padded K)
    __shared__ float    s_a[16][64];
    __shared__ float    a_a[16][64];
    __shared__ __align__(16) float lzj[16][64];
    __shared__ _Float16 hh[16][128];
    __shared__ float    outs[16][16];

    for (int idx = tid; idx < 16 * 64 * 8; idx += 256)
        (&zsb[0][0][0])[idx] = 0.f;
    __syncthreads();

    int wave = tid >> 5, lane = tid & 31;
    int m = lane & 15, kg = lane >> 4;
    float bfz = bf_z[0];

    for (int j = 0; j < 64; ++j) {
        // stage LZ[:, j, :] for this block's 16 batch rows (async b128/lane)
        {
            int br = tid >> 4, a = (tid & 15) * 4;    // 16 threads per row
            const float* g = LZ + ((size_t)(bb0 + br) * 64 + j) * 64 + a;
            unsigned l = (unsigned)(unsigned long long)&lzj[br][a];
            async_g2l_b128(l, g);
        }
        wait_asynccnt0();
        __syncthreads();

        // s[b][p] = tanh(T[b,p,:] - lzj[b,:]) . Wf_z + bf_z
#pragma unroll
        for (int rep = 0; rep < 4; ++rep) {
            int pp = tid + rep * 256;
            int b = pp >> 6, p = pp & 63;
            const float* tr = T + ((size_t)(bb0 + b) * 64 + p) * 64;
            float acc = 0.f;
            for (int a = 0; a < 64; ++a)
                acc += tanhf(tr[a] - lzj[b][a]) * Wf_z[a];
            s_a[b][p] = acc + bfz;
        }
        __syncthreads();

        // masked softmax over p (mask p < j)
        if (tid < 16) {
            int b = tid;
            float mx = -1e9f;
            for (int p = 0; p < j; ++p) mx = fmaxf(mx, s_a[b][p]);
            float sum = 0.f;
            for (int p = 0; p < 64; ++p) {
                float e = (p < j) ? expf(s_a[b][p] - mx) : 0.f;
                a_a[b][p] = e;
                sum += e;
            }
            float inv = 1.f / (sum + 1e-10f);
            for (int p = 0; p < 64; ++p) a_a[b][p] *= inv;
        }
        __syncthreads();

        // full = [xe_j | social_j | z_prev(72) | zero-pad] -> f16
        for (int idx = tid; idx < 16 * 224; idx += 256) {
            int b = idx / 224, d = idx % 224;
            int gb = bb0 + b;
            float v;
            if (d < 64) {
                v = x_enc[((size_t)gb * 64 + j) * 64 + d];
            } else if (d < 128) {
                v = social[((size_t)gb * 64 + j) * 64 + (d - 64)];
            } else if (d < 200) {
                int dd = d - 128;
                float acc = 0.f;
                if (dd < 64) {
                    for (int p = 0; p < 64; ++p)
                        acc += a_a[b][p] * x_enc[((size_t)gb * 64 + p) * 64 + dd];
                } else {
                    int q = dd - 64;
                    for (int p = 0; p < 64; ++p)
                        acc += a_a[b][p] * zsb[b][p][q];
                }
                v = acc;
            } else {
                v = 0.f;
            }
            fullh[b][d] = (_Float16)v;
        }
        __syncthreads();

        // MLP layer 1: (16x224)@(224x128) + b1, relu; wave = column tile
        {
            int n = wave * 16 + m;
            float8 c = {};
#pragma unroll
            for (int kb = 0; kb < 224; kb += 32) {
                half16 av, bv;
#pragma unroll
                for (int t = 0; t < 8; ++t) {
                    av[t]     = fullh[m][kb + kg * 16 + t];
                    av[8 + t] = fullh[m][kb + kg * 16 + 8 + t];
                }
#pragma unroll
                for (int t = 0; t < 16; ++t)
                    bv[t] = W1h[(kb + kg * 16 + t) * 128 + n];
                c = __builtin_amdgcn_wmma_f32_16x16x32_f16(false, av, false, bv, (short)0, c, false, false);
            }
            float biasv = b1[n];
#pragma unroll
            for (int v = 0; v < 8; ++v) {
                float r = c[v] + biasv;
                hh[v + 8 * kg][n] = (_Float16)fmaxf(r, 0.f);
            }
        }
        __syncthreads();

        // MLP layer 2: (16x128)@(128x16) + b2 on wave 0
        if (wave == 0) {
            float8 c = {};
#pragma unroll
            for (int kb = 0; kb < 128; kb += 32) {
                half16 av, bv;
#pragma unroll
                for (int t = 0; t < 8; ++t) {
                    av[t]     = hh[m][kb + kg * 16 + t];
                    av[8 + t] = hh[m][kb + kg * 16 + 8 + t];
                }
#pragma unroll
                for (int t = 0; t < 16; ++t)
                    bv[t] = W2h[(kb + kg * 16 + t) * 16 + m];
                c = __builtin_amdgcn_wmma_f32_16x16x32_f16(false, av, false, bv, (short)0, c, false, false);
            }
            float biasv = b2[m];
#pragma unroll
            for (int v = 0; v < 8; ++v)
                outs[v + 8 * kg][m] = c[v] + biasv;
        }
        __syncthreads();

        // reparameterize + write outputs + update zs buffer
        if (tid < 128) {
            int b = tid >> 3, d = tid & 7;
            int gr = (bb0 + b) * 64 + j;
            float mean = outs[b][d];
            float lg   = outs[b][d + 8];
            float z = eps[(size_t)gr * 8 + d] * expf(0.5f * lg) + mean;
            out_means[(size_t)gr * 8 + d] = mean;
            out_logs[(size_t)gr * 8 + d]  = lg;
            out_zs[(size_t)gr * 8 + d]    = z;
            zsb[b][j][d] = z;
        }
        __syncthreads();
    }
}

// ---------------------------------------------------------------------------
extern "C" void kernel_launch(void* const* d_in, const int* in_sizes, int n_in,
                              void* d_out, int out_size, void* d_ws, size_t ws_size,
                              hipStream_t stream)
{
    (void)in_sizes; (void)n_in; (void)out_size; (void)ws_size;

    const float* x_enc  = (const float*)d_in[0];
    const float* x_last = (const float*)d_in[1];
    /* d_in[2] = seq_start_end (unused: sequences are uniform P=64) */
    const float* eps  = (const float*)d_in[3];
    const float* We_x = (const float*)d_in[4];
    const float* be_x = (const float*)d_in[5];
    const float* Wl_x = (const float*)d_in[6];
    const float* bl_x = (const float*)d_in[7];
    const float* Wc_x = (const float*)d_in[8];
    const float* bc_x = (const float*)d_in[9];
    const float* Wf_x = (const float*)d_in[10];
    const float* bf_x = (const float*)d_in[11];
    const float* We_z = (const float*)d_in[12];
    const float* be_z = (const float*)d_in[13];
    const float* Wl_z = (const float*)d_in[14];
    const float* bl_z = (const float*)d_in[15];
    const float* Wf_z = (const float*)d_in[16];
    const float* bf_z = (const float*)d_in[17];
    const float* W1   = (const float*)d_in[18];
    const float* b1   = (const float*)d_in[19];
    const float* W2   = (const float*)d_in[20];
    const float* b2   = (const float*)d_in[21];

    const size_t NH = (size_t)NN * 64;   // 1,048,576 floats
    float* ws      = (float*)d_ws;
    float* E       = ws;
    float* Cc      = E + NH;
    float* EZ      = Cc + NH;
    float* LZ      = EZ + NH;
    float* T       = LZ + NH;
    float* alpha   = T + NH;             // B*P*P == NH
    float* social  = alpha + NH;
    float* keymask = social + NH;        // NN floats
    _Float16* W1h  = (_Float16*)(keymask + NN);
    _Float16* W2h  = W1h + 224 * 128;

    float* out_means = (float*)d_out;
    float* out_logs  = out_means + (size_t)NN * NDp;
    float* out_zs    = out_logs + (size_t)NN * NDp;

    wconv_kernel<<<8, 256, 0, stream>>>(W1, W2, W1h, W2h);
    gemm3_kernel<<<dim3(NN / 16, 3), 128, 0, stream>>>(
        x_enc, We_x, be_x, E, Wc_x, bc_x, Cc, We_z, be_z, EZ);
    prep_kernel<<<NN, 64, 0, stream>>>(x_enc, x_last, Wl_z, bl_z, EZ, LZ, T, keymask);
    score_kernel<<<NN, 32, 0, stream>>>(E, Cc, x_last, Wl_x, bl_x, Wf_x, bf_x, keymask, alpha);
    social_kernel<<<BB, 256, 0, stream>>>(alpha, x_enc, social);
    scan_kernel<<<BB / 16, 256, 0, stream>>>(x_enc, eps, T, LZ, Wf_z, bf_z, social,
                                             W1h, b1, W2h, b2,
                                             out_means, out_logs, out_zs);
}
